// SelfAttentionConv2dLite_69389491634589
// MI455X (gfx1250) — matrix operand order repaired and checked by reading, not compiled, not run on hardware
//
#include <hip/hip_runtime.h>
#include <hip/hip_bf16.h>

typedef __attribute__((ext_vector_type(16))) _Float16 v16h;
typedef __attribute__((ext_vector_type(8)))  _Float16 v8h;
typedef __attribute__((ext_vector_type(4)))  _Float16 v4h;
typedef __attribute__((ext_vector_type(8)))  float    v8f;

#define BATCH 8
#define CCH   256
#define NPIX  4096
#define ACH   16
#define BN    64    // m-columns per block
#define KT    32    // n (K) per WMMA tile step
#define PSTR  40    // pT row stride in halves (80 B: rows 16B-aligned)

// ---------------------------------------------------------------------------
// Kernel 1: qT[b][n][a] = sum_c Wq[a,c] * x[b,c,n] + bq[a]   (fp32 in ws)
// ---------------------------------------------------------------------------
__global__ __launch_bounds__(256) void qproj_kernel(
    const float* __restrict__ x, const float* __restrict__ Wq,
    const float* __restrict__ bq, float* __restrict__ qT)
{
    int b      = blockIdx.x >> 8;            // 256 n-blocks of 16 per batch
    int n_base = (blockIdx.x & 255) * 16;
    int t      = threadIdx.x;
    int a      = t & 15;
    int n      = n_base + (t >> 4);

    const float* xb = x + (size_t)b * CCH * NPIX + n;
    const float* w  = Wq + a * CCH;
    float acc = bq[a];
    #pragma unroll 8
    for (int c = 0; c < CCH; ++c)
        acc = fmaf(w[c], xb[(size_t)c * NPIX], acc);
    qT[((size_t)b * NPIX + n) * ACH + a] = acc;
}

// ---------------------------------------------------------------------------
// Kernel 2: x_h = (f16)x, flat copy. Removes all f32->f16 cvt from the hot loop.
// ---------------------------------------------------------------------------
__global__ __launch_bounds__(256) void xcast_kernel(
    const float* __restrict__ x, _Float16* __restrict__ x_h)
{
    size_t i = ((size_t)blockIdx.x * 256 + threadIdx.x) * 4;
    float4 f = *(const float4*)(x + i);
    v4h hv = {(_Float16)f.x, (_Float16)f.y, (_Float16)f.z, (_Float16)f.w};
    *(v4h*)(x_h + i) = hv;
}

// ---------------------------------------------------------------------------
// Kernel 3: flash-style attention. Block = 256 threads (8 waves), one batch b
// and 64 output columns m. Softmax over n (4096) per column m. PV product on
// v_wmma_f32_16x16x32_f16; each wave owns 32 rows of C. q-row tiles staged
// with global_load_async_to_lds_b64 (ASYNCcnt path).
// ---------------------------------------------------------------------------
__global__ __launch_bounds__(256) void attn_kernel(
    const float* __restrict__ x, const _Float16* __restrict__ x_h,
    const float* __restrict__ qT, const float* __restrict__ scale_p,
    float* __restrict__ out)
{
    const int b      = blockIdx.x >> 6;       // 64 m-blocks per batch
    const int m_base = (blockIdx.x & 63) * BN;
    const int t      = threadIdx.x;
    const int lane   = t & 31;
    const int wave   = t >> 5;
    const int lh     = lane & 15;
    const float scale = scale_p[0];

    __shared__ float    qm[BN][ACH];       // q columns for this m-tile (4 KB)
    __shared__ float    qn[KT][ACH];       // q rows for current n-tile (2 KB)
    __shared__ _Float16 pT[BN][PSTR];      // P tile, B-layout-transposed (5 KB)
    __shared__ float    red_mx[4 * BN];
    __shared__ float    red_sm[4 * BN];
    __shared__ float    colmax[BN];
    __shared__ float    colden[BN];

    const float* qTb = qT + (size_t)b * NPIX * ACH;

    // stage q[:, m] for the block's 64 columns
    for (int i = t; i < BN * ACH; i += 256)
        qm[i >> 4][i & 15] = qTb[(size_t)(m_base + (i >> 4)) * ACH + (i & 15)];
    __syncthreads();

    // ---- pass 1: online (max, sum-exp) over n, 4 threads per column m ----
    {
        const int m_local = t & 63;
        const int chunk   = t >> 6;       // 0..3, each covers 1024 n
        float mq[ACH];
        #pragma unroll
        for (int a = 0; a < ACH; ++a) mq[a] = qm[m_local][a];

        float mx = -3.0e38f, sm = 0.0f;
        const int n0 = chunk * 1024;
        for (int n = n0; n < n0 + 1024; ++n) {
            const float* qp = qTb + (size_t)n * ACH;
            float s = 0.0f;
            #pragma unroll
            for (int a = 0; a < ACH; ++a) s = fmaf(qp[a], mq[a], s);
            s *= scale;
            float nm = fmaxf(mx, s);
            sm = sm * __expf(mx - nm) + __expf(s - nm);
            mx = nm;
        }
        red_mx[chunk * BN + m_local] = mx;
        red_sm[chunk * BN + m_local] = sm;
    }
    __syncthreads();
    if (t < BN) {
        float mx = red_mx[t], sm = red_sm[t];
        #pragma unroll
        for (int k = 1; k < 4; ++k) {
            float m2 = red_mx[k * BN + t], s2 = red_sm[k * BN + t];
            float nm = fmaxf(mx, m2);
            sm = sm * __expf(mx - nm) + s2 * __expf(m2 - nm);
            mx = nm;
        }
        colmax[t] = mx;
        colden[t] = sm;
    }

    // ---- pass 2: PV accumulation with WMMA ----
    const int c_wbase = wave * 32;                 // this wave's 32 C-rows
    const int ks0A = (lane < 16) ? 0 : 8;          // A per-lane K base (interleaved halves)
    const int ks0B = (lane < 16) ? 0 : 8;          // B per-lane K base (interleaved halves)
    const _Float16* xh0 =
        x_h + ((size_t)b * CCH + (c_wbase + lh)) * NPIX;            // row for i=0
    const _Float16* xh1 = xh0 + (size_t)16 * NPIX;                  // row for i=1
    v8f acc[2][4] = {};

    // P producer mapping: thread t -> column m = t>>2, 8 consecutive K.
    const int pm  = t >> 2;
    const int pk0 = (t & 3) * 8;

    // async copy setup: thread t moves bytes [8t, 8t+8) of the 2 KB qn tile
    const unsigned   qn_lds  = (unsigned)(uintptr_t)(&qn[0][0]) + (unsigned)(t * 8);
    unsigned long long qn_ga = (unsigned long long)(uintptr_t)qTb
                               + (unsigned long long)(t * 8);

    for (int n_base = 0; n_base < NPIX; n_base += KT) {
        __syncthreads();   // pT/qn consumers of previous iteration are done
        // stage 32 q-rows (2 KB) via async global->LDS, b64 per lane
        asm volatile("global_load_async_to_lds_b64 %0, %1, off"
                     :: "v"(qn_lds), "v"(qn_ga) : "memory");
        qn_ga += (unsigned long long)(KT * ACH * 4);
        asm volatile("s_wait_asynccnt 0" ::: "memory");
        __syncthreads();

        // build P tile directly in f16 B-operand layout: pT[m][k]
        {
            const float cm = colmax[pm];
            v8h pv8;
            #pragma unroll
            for (int j = 0; j < 8; ++j) {
                const int k = pk0 + j;
                float s = 0.0f;
                #pragma unroll
                for (int a = 0; a < ACH; ++a) s = fmaf(qn[k][a], qm[pm][a], s);
                pv8[j] = (_Float16)__expf(s * scale - cm);
            }
            *(v8h*)&pT[pm][pk0] = pv8;   // 16-byte aligned LDS store
        }
        __syncthreads();

        // B operands: two 16-byte runs (K = ks0B..+7, ks0B+16..+23) per (lane, m-tile)
        v16h pv[4];
        #pragma unroll
        for (int j = 0; j < 4; ++j) {
            v8h lo = *(const v8h*)&pT[j * 16 + lh][ks0B];
            v8h hi = *(const v8h*)&pT[j * 16 + lh][ks0B + 16];
            pv[j] = __builtin_shufflevector(lo, hi, 0, 1, 2, 3, 4, 5, 6, 7,
                                                    8, 9, 10, 11, 12, 13, 14, 15);
        }

        // A operands: two b128 loads of pre-converted halves, then 8 WMMAs
        #pragma unroll
        for (int i = 0; i < 2; ++i) {
            const _Float16* xr = (i == 0 ? xh0 : xh1) + n_base;
            v8h alo = *(const v8h*)&xr[ks0A];        // K = ks0A .. ks0A+7
            v8h ahi = *(const v8h*)&xr[ks0A + 16];   // K = ks0A+16 .. ks0A+23
            v16h av = __builtin_shufflevector(alo, ahi, 0, 1, 2, 3, 4, 5, 6, 7,
                                                        8, 9, 10, 11, 12, 13, 14, 15);
            #pragma unroll
            for (int j = 0; j < 4; ++j)
                acc[i][j] = __builtin_amdgcn_wmma_f32_16x16x32_f16(
                    false, av, false, pv[j], (short)0, acc[i][j], false, false);
        }
    }
    __syncthreads();

    // ---- epilogue: out = acc/den + x (residual) ----
    #pragma unroll
    for (int i = 0; i < 2; ++i) {
        #pragma unroll
        for (int j = 0; j < 4; ++j) {
            const int m_local = j * 16 + lh;
            const float inv = 1.0f / colden[m_local];
            const int m = m_base + m_local;
            #pragma unroll
            for (int r = 0; r < 8; ++r) {
                const int cr = c_wbase + i * 16 + ((lane < 16) ? r : r + 8);
                const size_t idx = ((size_t)b * CCH + cr) * NPIX + m;
                out[idx] = acc[i][j][r] * inv + x[idx];
            }
        }
    }
}

extern "C" void kernel_launch(void* const* d_in, const int* in_sizes, int n_in,
                              void* d_out, int out_size, void* d_ws, size_t ws_size,
                              hipStream_t stream) {
    const float* x     = (const float*)d_in[0];
    const float* Wq    = (const float*)d_in[1];
    const float* bq    = (const float*)d_in[2];
    const float* scale = (const float*)d_in[3];
    float* out = (float*)d_out;

    float*    qT  = (float*)d_ws;                             // [B][N][16] fp32, 2 MB
    _Float16* x_h = (_Float16*)((char*)d_ws + (size_t)BATCH * NPIX * ACH * 4);

    qproj_kernel<<<BATCH * (NPIX / 16), 256, 0, stream>>>(x, Wq, bq, qT);
    xcast_kernel<<<(BATCH * CCH * NPIX) / 1024, 256, 0, stream>>>(x, x_h);
    attn_kernel<<<BATCH * (NPIX / BN), 256, 0, stream>>>(x, x_h, qT, scale, out);
}
